// PerformerAttention_18889266168449
// MI455X (gfx1250) — matrix operand order repaired and checked by reading, not compiled, not run on hardware
//
#include <hip/hip_runtime.h>
#include <hip/hip_bf16.h>
#include <math.h>

// ---------------- problem constants ----------------
#define B_      8
#define N_      4096
#define C_      768
#define H_      12
#define HD_     64
#define BH_     96          // B_*H_
#define BN_     32768       // B_*N_
#define C3_     2304        // 3*C_
#define MREAL_  266         // m = hd*log(hd)
#define MP_     288         // m padded to 9*32 for WMMA K-slices
#define NORM_   0.35355339059327373f   // 1/64^0.25
#define EPS_    1e-4f

typedef __attribute__((ext_vector_type(16))) _Float16 v16h;
typedef __attribute__((ext_vector_type(8)))  _Float16 v8h;
typedef __attribute__((ext_vector_type(8)))  float    v8f;

static __device__ __forceinline__ int lane_id() { return threadIdx.x & 31; }

// ---- WMMA fragment loaders (16x16x32 f16, wave32; layouts per CDNA5 ISA 7.12.2) ----
// A (16x32, MxK), row-major LDS source: two contiguous 16B runs per lane.
static __device__ __forceinline__ v16h frag_a_row(const _Float16* p, int ld) {
  int l = lane_id(); int r = l & 15, g = l >> 4;
  const _Float16* q = p + r * ld + g * 8;
  v8h lo = *(const v8h*)q;
  v8h hi = *(const v8h*)(q + 16);
  return __builtin_shufflevector(lo, hi, 0,1,2,3,4,5,6,7,8,9,10,11,12,13,14,15);
}
// B (32x16, KxN) with LDS source storing B^T row-major: two contiguous 16B runs per lane.
static __device__ __forceinline__ v16h frag_b_colmaj(const _Float16* p, int ld) {
  int l = lane_id(); int n = l & 15, g = l >> 4;
  const _Float16* q = p + n * ld + g * 8;
  v8h lo = *(const v8h*)q;
  v8h hi = *(const v8h*)(q + 16);
  return __builtin_shufflevector(lo, hi, 0,1,2,3,4,5,6,7,8,9,10,11,12,13,14,15);
}
static __device__ __forceinline__ v8f wmma_f16(v16h a, v16h b, v8f c) {
  return __builtin_amdgcn_wmma_f32_16x16x32_f16(false, a, false, b, (short)0, c, false, false);
}
// order-preserving float<->uint for atomic max over signed floats
static __device__ __forceinline__ unsigned f2ord(float f) {
  unsigned u = __float_as_uint(f);
  return (u & 0x80000000u) ? ~u : (u | 0x80000000u);
}
static __device__ __forceinline__ float ord2f(unsigned u) {
  return __uint_as_float((u & 0x80000000u) ? (u ^ 0x80000000u) : ~u);
}

// ---- CDNA5 async global->LDS copy (16B per lane, tracked by ASYNCcnt) ----
static __device__ __forceinline__ void async_copy16(const void* g, void* lds) {
  unsigned loff = (unsigned)(size_t)lds;     // flat LDS addr low 32 bits = LDS byte offset
  asm volatile("global_load_async_to_lds_b128 %0, %1, off"
               :: "v"(loff), "v"(g) : "memory");
}
#define ASYNC_WAIT(n) asm volatile("s_wait_asynccnt " #n ::: "memory")

// ---------------- prep kernels ----------------
__global__ void k_f32_to_f16(const float* __restrict__ s, _Float16* __restrict__ d, int n) {
  for (int i = blockIdx.x * blockDim.x + threadIdx.x; i < n; i += gridDim.x * blockDim.x)
    d[i] = (_Float16)s[i];
}
// d[c*rows + r] = (f16)s[r*cols + c]  (transpose + convert; one-time weight prep)
__global__ void k_trans_f16(const float* __restrict__ s, _Float16* __restrict__ d,
                            int rows, int cols) {
  int total = rows * cols;
  for (int i = blockIdx.x * blockDim.x + threadIdx.x; i < total; i += gridDim.x * blockDim.x) {
    int c = i / rows, r = i % rows;
    d[i] = (_Float16)s[(size_t)r * cols + c];
  }
}
__global__ void k_pad_proj(const float* __restrict__ proj, _Float16* __restrict__ ph) {
  int i = blockIdx.x * blockDim.x + threadIdx.x;
  if (i < MP_ * HD_) {
    int m = i / HD_;
    ph[i] = (m < MREAL_) ? (_Float16)proj[i] : (_Float16)0.f;
  }
}
__global__ void k_zero_u32(unsigned* __restrict__ p, int n) {
  for (int i = blockIdx.x * blockDim.x + threadIdx.x; i < n; i += gridDim.x * blockDim.x)
    p[i] = 0u;
}

// ---------------- qkv GEMM: [BN,768] x [768,2304] -> scaled f16 q/k + v ----------------
// 64x128 block tile, double-buffered LDS, fully async staged (weights pre-transposed).
__global__ void k_gemm_qkv(const _Float16* __restrict__ xh, const _Float16* __restrict__ whT,
                           _Float16* __restrict__ qn, _Float16* __restrict__ kn,
                           _Float16* __restrict__ vh) {
  __shared__ __align__(16) _Float16 Ab[2][64 * 32];    // A row-major
  __shared__ __align__(16) _Float16 Bb[2][128 * 32];   // B^T row-major: Bt[c][k]
  int tid = threadIdx.x, w = tid >> 5;
  int r0 = blockIdx.x * 64, c0 = blockIdx.y * 128;
  int ti = w >> 1, tjb = (w & 1) * 4;
  v8f acc[4] = {};
  int arow = tid >> 2, part = tid & 3;
  int bc1 = tid >> 2, bc2 = bc1 + 64;
  auto issue = [&](int kt2, int nb) {
    int k0 = kt2 * 32;
    async_copy16(xh + (size_t)(r0 + arow) * C_ + k0 + part * 8, Ab[nb] + arow * 32 + part * 8);
    async_copy16(whT + (size_t)(c0 + bc1) * C_ + k0 + part * 8, Bb[nb] + bc1 * 32 + part * 8);
    async_copy16(whT + (size_t)(c0 + bc2) * C_ + k0 + part * 8, Bb[nb] + bc2 * 32 + part * 8);
  };
  issue(0, 0);
  for (int kt = 0; kt < 24; ++kt) {
    int cur = kt & 1;
    if (kt + 1 < 24) { issue(kt + 1, cur ^ 1); ASYNC_WAIT(3); }
    else             { ASYNC_WAIT(0); }
    __syncthreads();
    v16h a = frag_a_row(Ab[cur] + ti * 16 * 32, 32);
#pragma unroll
    for (int j = 0; j < 4; ++j)
      acc[j] = wmma_f16(a, frag_b_colmaj(Bb[cur] + (tjb + j) * 16 * 32, 32), acc[j]);
    __syncthreads();
  }
  int nl = lane_id() & 15, g = lane_id() >> 4;
#pragma unroll
  for (int j = 0; j < 4; ++j) {
    int gc = c0 + (tjb + j) * 16 + nl;
    int which = gc / C_, rem = gc % C_, h = rem >> 6, d = rem & 63;
#pragma unroll
    for (int i = 0; i < 8; ++i) {
      int gr = r0 + ti * 16 + i + 8 * g;
      int b = gr >> 12, nr = gr & (N_ - 1);
      size_t di = ((size_t)(b * H_ + h) * N_ + nr) * HD_ + d;
      float v = acc[j][i];
      if (which == 0)      qn[di] = (_Float16)(v * NORM_);
      else if (which == 1) kn[di] = (_Float16)(v * NORM_);
      else                 vh[di] = (_Float16)v;
    }
  }
}

// ---------------- per-row 0.5*||.||^2 for normalized q,k ----------------
__global__ void k_diag(const _Float16* __restrict__ qn, const _Float16* __restrict__ kn,
                       float* __restrict__ dq, float* __restrict__ dk) {
  int i = blockIdx.x * blockDim.x + threadIdx.x;
  if (i >= BH_ * N_) return;
  const v8h* q = (const v8h*)(qn + (size_t)i * HD_);
  const v8h* k = (const v8h*)(kn + (size_t)i * HD_);
  float sq = 0.f, sk = 0.f;
#pragma unroll
  for (int c = 0; c < 8; ++c) {
    v8h a = q[c], b = k[c];
#pragma unroll
    for (int j = 0; j < 8; ++j) {
      float fa = (float)a[j]; sq += fa * fa;
      float fb = (float)b[j]; sk += fb * fb;
    }
  }
  dq[i] = 0.5f * sq;
  dk[i] = 0.5f * sk;
}

// ---------------- pass 1 over k_dash: global max via ordered-uint atomic ----------------
__global__ void k_kmax(const _Float16* __restrict__ kn, const _Float16* __restrict__ ph,
                       unsigned* __restrict__ kmax) {
  __shared__ __align__(16) _Float16 Ks[64 * 64];
  __shared__ __align__(16) _Float16 Ps[MP_ * HD_];
  __shared__ float wred[8];
  int tid = threadIdx.x, w = tid >> 5;
  int bh = blockIdx.x, n0 = blockIdx.y * 64;
  const _Float16* kb = kn + ((size_t)bh * N_ + n0) * HD_;
  async_copy16(kb + tid * 8, Ks + tid * 8);
  async_copy16(kb + (tid + 256) * 8, Ks + (tid + 256) * 8);
  for (int c = tid; c < MP_ * HD_ / 8; c += 256) async_copy16(ph + c * 8, Ps + c * 8);
  ASYNC_WAIT(0);
  __syncthreads();
  float lm = -3.0e38f;
  int nl = lane_id() & 15;
  int ti = w & 3, tjb = (w >> 2) * 9;
  v16h a0 = frag_a_row(Ks + ti * 16 * 64, 64);
  v16h a1 = frag_a_row(Ks + ti * 16 * 64 + 32, 64);
  for (int tj = tjb; tj < tjb + 9; ++tj) {
    v8f acc = {};
    acc = wmma_f16(a0, frag_b_colmaj(Ps + tj * 16 * 64, 64), acc);
    acc = wmma_f16(a1, frag_b_colmaj(Ps + tj * 16 * 64 + 32, 64), acc);
    int m = tj * 16 + nl;
    if (m < MREAL_) {
#pragma unroll
      for (int i = 0; i < 8; ++i) lm = fmaxf(lm, acc[i]);
    }
  }
  for (int off = 16; off; off >>= 1) lm = fmaxf(lm, __shfl_down(lm, off));
  if (lane_id() == 0) wred[w] = lm;
  __syncthreads();
  if (tid == 0) {
    float m = wred[0];
    for (int i = 1; i < 8; ++i) m = fmaxf(m, wred[i]);
    atomicMax(kmax, f2ord(m));
  }
}

// ---------------- pass 2: k features + kv/ksum contraction (k1 never leaves LDS) ----------------
__global__ void k_kfeat_kv(const _Float16* __restrict__ kn, const _Float16* __restrict__ vh,
                           const _Float16* __restrict__ ph, const float* __restrict__ dkg,
                           const unsigned* __restrict__ kmax, float* __restrict__ kvacc,
                           float* __restrict__ ksum) {
  extern __shared__ char smem[];
  _Float16* Ks  = (_Float16*)smem;          // 64 x 64   (rows n, cols d)
  _Float16* Vt  = Ks + 64 * 64;             // 64 x 64   V transposed: Vt[d][n]
  _Float16* Ps  = Vt + 64 * 64;             // 288 x 64
  _Float16* K1t = Ps + MP_ * HD_;           // 288 x 64  features transposed: K1t[m][n]
  float*    dg  = (float*)(K1t + MP_ * 64); // 64
  int tid = threadIdx.x, w = tid >> 5;
  int bh = blockIdx.x, n0 = blockIdx.y * 64;
  const _Float16* kb = kn + ((size_t)bh * N_ + n0) * HD_;
  const _Float16* vb = vh + ((size_t)bh * N_ + n0) * HD_;
  async_copy16(kb + tid * 8, Ks + tid * 8);
  async_copy16(kb + (tid + 256) * 8, Ks + (tid + 256) * 8);
  for (int c = tid; c < MP_ * HD_ / 8; c += 256) async_copy16(ph + c * 8, Ps + c * 8);
  for (int c = tid; c < 64 * 64 / 8; c += 256) {   // transpose V into LDS
    int n = c >> 3, d0 = (c & 7) * 8;
    v8h val = ((const v8h*)vb)[c];
#pragma unroll
    for (int j = 0; j < 8; ++j) Vt[(d0 + j) * 64 + n] = val[j];
  }
  if (tid < 64) dg[tid] = dkg[(size_t)bh * N_ + n0 + tid];
  float stab = ord2f(*kmax);
  ASYNC_WAIT(0);
  __syncthreads();
  int nl = lane_id() & 15, g = lane_id() >> 4;
  // k_dash tiles -> exp -> K1t (padded cols forced to 0); A frags hoisted per wave
  {
    int ti = w & 3, tjb = (w >> 2) * 9;
    v16h a0 = frag_a_row(Ks + ti * 16 * 64, 64);
    v16h a1 = frag_a_row(Ks + ti * 16 * 64 + 32, 64);
    for (int tj = tjb; tj < tjb + 9; ++tj) {
      v8f acc = {};
      acc = wmma_f16(a0, frag_b_colmaj(Ps + tj * 16 * 64, 64), acc);
      acc = wmma_f16(a1, frag_b_colmaj(Ps + tj * 16 * 64 + 32, 64), acc);
      int m = tj * 16 + nl;
#pragma unroll
      for (int i = 0; i < 8; ++i) {
        int r = ti * 16 + i + 8 * g;
        float val = (m < MREAL_) ? (__expf(acc[i] - dg[r] - stab) + EPS_) : 0.f;
        K1t[m * 64 + r] = (_Float16)val;
      }
    }
  }
  __syncthreads();
  // ksum partial (contiguous reduction along n)
  for (int m = tid; m < MP_; m += 256) {
    float s = 0.f;
    for (int r = 0; r < 64; ++r) s += (float)K1t[m * 64 + r];
    atomicAdd(&ksum[bh * MP_ + m], s);
  }
  // kv partial: [288 x 64] = K1t (A row-major) * V (via Vt colmaj), K = 64.
  // All waves share the same 8 B fragments -> preload into registers.
  float* kvb = kvacc + (size_t)bh * MP_ * HD_;
  v16h vb0[4], vb1[4];
#pragma unroll
  for (int tj = 0; tj < 4; ++tj) {
    vb0[tj] = frag_b_colmaj(Vt + tj * 16 * 64, 64);
    vb1[tj] = frag_b_colmaj(Vt + tj * 16 * 64 + 32, 64);
  }
  for (int ti = w; ti < 18; ti += 8) {
    v16h a0 = frag_a_row(K1t + ti * 16 * 64, 64);
    v16h a1 = frag_a_row(K1t + ti * 16 * 64 + 32, 64);
#pragma unroll
    for (int tj = 0; tj < 4; ++tj) {
      v8f acc = {};
      acc = wmma_f16(a0, vb0[tj], acc);
      acc = wmma_f16(a1, vb1[tj], acc);
#pragma unroll
      for (int i = 0; i < 8; ++i) {
        int m = ti * 16 + i + 8 * g, d = tj * 16 + nl;
        atomicAdd(&kvb[m * HD_ + d], acc[i]);
      }
    }
  }
}

// ---------------- q features + out = (q1 @ kv) / (q1 . ksum), fused ----------------
__global__ void k_qfeat_out(const _Float16* __restrict__ qn, const _Float16* __restrict__ ph,
                            const float* __restrict__ dqg, const float* __restrict__ kvg,
                            const float* __restrict__ ksg, _Float16* __restrict__ outpre) {
  extern __shared__ char smem[];
  _Float16* Qs    = (_Float16*)smem;          // 32 x 64
  _Float16* Ps    = Qs + 32 * 64;             // 288 x 64
  _Float16* KVt   = Ps + MP_ * HD_;           // 64 x 288  kv transposed: KVt[d][m]
  _Float16* Q1    = KVt + HD_ * MP_;          // 32 x 288  (row-major A)
  float*    dashs = (float*)(Q1 + 32 * MP_);  // 32 x 288
  float*    kss   = dashs + 32 * MP_;         // 288
  float*    dg    = kss + MP_;                // 32
  float*    zr    = dg + 32;                  // 32
  int tid = threadIdx.x, w = tid >> 5;
  int bh = blockIdx.x, n0 = blockIdx.y * 32;
  int b = bh / H_, h = bh % H_;
  const _Float16* qb = qn + ((size_t)bh * N_ + n0) * HD_;
  const float* kvb = kvg + (size_t)bh * MP_ * HD_;
  async_copy16(qb + tid * 8, Qs + tid * 8);
  for (int c = tid; c < MP_ * HD_ / 8; c += 256) async_copy16(ph + c * 8, Ps + c * 8);
  for (int t = tid; t < MP_ * HD_; t += 256) {   // transpose + f32->f16 convert
    int m = t >> 6, d = t & 63;
    KVt[d * MP_ + m] = (_Float16)kvb[t];
  }
  for (int t = tid; t < MP_; t += 256) kss[t] = ksg[bh * MP_ + t];
  if (tid < 32) dg[tid] = dqg[(size_t)bh * N_ + n0 + tid];
  ASYNC_WAIT(0);
  __syncthreads();
  int nl = lane_id() & 15, g = lane_id() >> 4;
  // q_dash tiles (2 x 18) -> fp32 LDS; A frags hoisted per wave
  {
    int ti = w >> 2, tj0 = w & 3;
    v16h a0 = frag_a_row(Qs + ti * 16 * 64, 64);
    v16h a1 = frag_a_row(Qs + ti * 16 * 64 + 32, 64);
    for (int tj = tj0; tj < 18; tj += 4) {
      v8f acc = {};
      acc = wmma_f16(a0, frag_b_colmaj(Ps + tj * 16 * 64, 64), acc);
      acc = wmma_f16(a1, frag_b_colmaj(Ps + tj * 16 * 64 + 32, 64), acc);
#pragma unroll
      for (int i = 0; i < 8; ++i)
        dashs[(ti * 16 + i + 8 * g) * MP_ + tj * 16 + nl] = acc[i];
    }
  }
  __syncthreads();
  // per-row stab, exp -> Q1, and z = 1/(q1 . ksum)
  if (tid < 32) {
    int r = tid;
    float mx = -3.0e38f;
    for (int m = 0; m < MREAL_; ++m) mx = fmaxf(mx, dashs[r * MP_ + m]);
    float dgr = dg[r], s = 0.f;
    for (int m = 0; m < MP_; ++m) {
      float val = (m < MREAL_) ? (__expf(dashs[r * MP_ + m] - dgr - mx) + EPS_) : 0.f;
      Q1[r * MP_ + m] = (_Float16)val;
      s += val * kss[m];
    }
    zr[r] = 1.f / s;
  }
  __syncthreads();
  // out tiles: 2 x 4 (one per wave), K = 288 = 9 slices of 32
  {
    int ti = w >> 2, tj = w & 3;
    v8f acc = {};
#pragma unroll
    for (int kk = 0; kk < 9; ++kk)
      acc = wmma_f16(frag_a_row(Q1 + ti * 16 * MP_ + kk * 32, MP_),
                     frag_b_colmaj(KVt + tj * 16 * MP_ + kk * 32, MP_), acc);
#pragma unroll
    for (int i = 0; i < 8; ++i) {
      int r = ti * 16 + i + 8 * g;
      int n = n0 + r, d = tj * 16 + nl;
      outpre[((size_t)(b * N_ + n)) * C_ + h * HD_ + d] = (_Float16)(acc[i] * zr[r]);
    }
  }
}

// ---------------- final GEMM: outpre [BN,768] x W_proj [768,768] + bias -> fp32 ----------------
__global__ void k_gemm_out(const _Float16* __restrict__ oh, const _Float16* __restrict__ wphT,
                           const float* __restrict__ bias, float* __restrict__ y) {
  __shared__ __align__(16) _Float16 Ab[2][64 * 32];
  __shared__ __align__(16) _Float16 Bb[2][128 * 32];
  int tid = threadIdx.x, w = tid >> 5;
  int r0 = blockIdx.x * 64, c0 = blockIdx.y * 128;
  int ti = w >> 1, tjb = (w & 1) * 4;
  v8f acc[4] = {};
  int arow = tid >> 2, part = tid & 3;
  int bc1 = tid >> 2, bc2 = bc1 + 64;
  auto issue = [&](int kt2, int nb) {
    int k0 = kt2 * 32;
    async_copy16(oh + (size_t)(r0 + arow) * C_ + k0 + part * 8, Ab[nb] + arow * 32 + part * 8);
    async_copy16(wphT + (size_t)(c0 + bc1) * C_ + k0 + part * 8, Bb[nb] + bc1 * 32 + part * 8);
    async_copy16(wphT + (size_t)(c0 + bc2) * C_ + k0 + part * 8, Bb[nb] + bc2 * 32 + part * 8);
  };
  issue(0, 0);
  for (int kt = 0; kt < 24; ++kt) {
    int cur = kt & 1;
    if (kt + 1 < 24) { issue(kt + 1, cur ^ 1); ASYNC_WAIT(3); }
    else             { ASYNC_WAIT(0); }
    __syncthreads();
    v16h a = frag_a_row(Ab[cur] + ti * 16 * 32, 32);
#pragma unroll
    for (int j = 0; j < 4; ++j)
      acc[j] = wmma_f16(a, frag_b_colmaj(Bb[cur] + (tjb + j) * 16 * 32, 32), acc[j]);
    __syncthreads();
  }
  int nl = lane_id() & 15, g = lane_id() >> 4;
#pragma unroll
  for (int j = 0; j < 4; ++j) {
    int gc = c0 + (tjb + j) * 16 + nl;
    float bv = bias[gc];
#pragma unroll
    for (int i = 0; i < 8; ++i) {
      int gr = r0 + ti * 16 + i + 8 * g;
      y[(size_t)gr * C_ + gc] = acc[j][i] + bv;
    }
  }
}

// ---------------- host ----------------
extern "C" void kernel_launch(void* const* d_in, const int* in_sizes, int n_in,
                              void* d_out, int out_size, void* d_ws, size_t ws_size,
                              hipStream_t stream) {
  (void)in_sizes; (void)n_in; (void)out_size; (void)ws_size;
  const float* x     = (const float*)d_in[0];
  const float* Wqkv  = (const float*)d_in[1];
  const float* Wproj = (const float*)d_in[2];
  const float* bproj = (const float*)d_in[3];
  const float* projm = (const float*)d_in[4];

  char* ws = (char*)d_ws;
  size_t off = 0;
  auto alloc = [&](size_t bytes) { size_t o = off; off += (bytes + 255) & ~(size_t)255; return o; };
  size_t o_xh  = alloc((size_t)BN_ * C_ * 2);
  size_t o_wh  = alloc((size_t)C_ * C3_ * 2);            // W_qkv^T (f16)
  size_t o_ph  = alloc((size_t)MP_ * HD_ * 2);
  size_t o_wph = alloc((size_t)C_ * C_ * 2);             // W_proj^T (f16)
  size_t o_qn  = alloc((size_t)BH_ * N_ * HD_ * 2);
  size_t o_kn  = alloc((size_t)BH_ * N_ * HD_ * 2);
  size_t o_vh  = alloc((size_t)BH_ * N_ * HD_ * 2);
  size_t o_dq  = alloc((size_t)BH_ * N_ * 4);
  size_t o_dk  = alloc((size_t)BH_ * N_ * 4);
  size_t o_kv  = alloc((size_t)BH_ * MP_ * HD_ * 4);     // contiguous zero region start
  size_t o_ks  = alloc((size_t)BH_ * MP_ * 4);
  size_t o_km  = alloc(256);                             // ordered-uint global max
  size_t o_op  = alloc((size_t)BN_ * C_ * 2);

  _Float16* xh   = (_Float16*)(ws + o_xh);
  _Float16* whT  = (_Float16*)(ws + o_wh);
  _Float16* ph   = (_Float16*)(ws + o_ph);
  _Float16* wphT = (_Float16*)(ws + o_wph);
  _Float16* qn   = (_Float16*)(ws + o_qn);
  _Float16* kn   = (_Float16*)(ws + o_kn);
  _Float16* vh   = (_Float16*)(ws + o_vh);
  float*    dq   = (float*)(ws + o_dq);
  float*    dk   = (float*)(ws + o_dk);
  float*    kv   = (float*)(ws + o_kv);
  float*    ksm  = (float*)(ws + o_ks);
  unsigned* km   = (unsigned*)(ws + o_km);
  _Float16* op   = (_Float16*)(ws + o_op);

  // prep: f16 staging, weight pre-transpose, zero accumulators (+ ordered-0 max sentinel)
  k_f32_to_f16<<<2048, 256, 0, stream>>>(x, xh, BN_ * C_);
  k_trans_f16<<<2048, 256, 0, stream>>>(Wqkv, whT, C_, C3_);
  k_trans_f16<<<1024, 256, 0, stream>>>(Wproj, wphT, C_, C_);
  k_pad_proj<<<(MP_ * HD_ + 255) / 256, 256, 0, stream>>>(projm, ph);
  int zwords = (int)((o_km + 256 - o_kv) / 4);
  k_zero_u32<<<2048, 256, 0, stream>>>((unsigned*)(ws + o_kv), zwords);

  // qkv projection GEMM (64x128 tiles, async double-buffered)
  k_gemm_qkv<<<dim3(BN_ / 64, C3_ / 128), 256, 0, stream>>>(xh, whT, qn, kn, vh);
  // diag terms
  k_diag<<<(BH_ * N_ + 255) / 256, 256, 0, stream>>>(qn, kn, dq, dk);
  // k_dash global max (pass 1)
  k_kmax<<<dim3(BH_, N_ / 64), 256, 0, stream>>>(kn, ph, km);
  // k features + kv/ksum (pass 2)
  size_t shm_k = (size_t)(64 * 64 + 64 * 64 + MP_ * HD_ + MP_ * 64) * sizeof(_Float16)
               + 64 * sizeof(float);
  k_kfeat_kv<<<dim3(BH_, N_ / 64), 256, shm_k, stream>>>(kn, vh, ph, dk, km, kv, ksm);
  // q features + normalized output
  size_t shm_q = (size_t)(32 * 64 + MP_ * HD_ + HD_ * MP_ + 32 * MP_) * sizeof(_Float16)
               + (size_t)(32 * MP_ + MP_ + 32 + 32) * sizeof(float);
  k_qfeat_out<<<dim3(BH_, N_ / 32), 256, shm_q, stream>>>(qn, ph, dq, kv, ksm, op);
  // final projection + bias -> fp32 d_out
  k_gemm_out<<<dim3(BN_ / 64, C_ / 128), 256, 0, stream>>>(op, wphT, bproj, (float*)d_out);
}
